// OutputBlock_19146964206348
// MI455X (gfx1250) — matrix operand order repaired and checked by reading, not compiled, not run on hardware
//
#include <hip/hip_runtime.h>
#include <hip/hip_bf16.h>

#define HIDDEN     128
#define OUT_EMB    256
#define OUT_CH     128
#define NUM_RADIAL 6
#define TILE_M     256          // nodes per workgroup (8 waves x 32 rows)
#define LDS_STRIDE 256          // activation row stride (elements)

// bf16 weight blob offsets (elements) inside workspace
#define OFF_PROJ 0
#define OFF_W0   (OFF_PROJ + OUT_EMB * HIDDEN)
#define OFF_W1   (OFF_W0   + OUT_EMB * OUT_EMB)
#define OFF_W2   (OFF_W1   + OUT_EMB * OUT_EMB)
#define OFF_FIN  (OFF_W2   + OUT_EMB * OUT_EMB)

typedef __attribute__((ext_vector_type(16))) __bf16 v16bf;
typedef __attribute__((ext_vector_type(8)))  __bf16 v8bf;
typedef __attribute__((ext_vector_type(8)))  float  v8f;

// ---------------------------------------------------------------------------
// Kernel 0: zero the scatter accumulator
// ---------------------------------------------------------------------------
__global__ void zero_f32_kernel(float* __restrict__ p, int n) {
    int i = blockIdx.x * blockDim.x + threadIdx.x;
    if (i < n) p[i] = 0.0f;
}

// Kernel 0b: one-shot f32 -> bf16 weight conversion into workspace
__global__ void cvt_bf16_kernel(const float* __restrict__ src,
                                __bf16* __restrict__ dst, int n) {
    int i = blockIdx.x * blockDim.x + threadIdx.x;
    if (i < n) dst[i] = (__bf16)src[i];
}

// ---------------------------------------------------------------------------
// Kernel 1: edge phase (memory bound).  basis = rbf @ Wb^T ; msg = basis * x ;
// f32 atomic scatter into agg[N,128] (25.6 MB -> L2 resident).
// ---------------------------------------------------------------------------
__global__ __launch_bounds__(256)
void edge_scatter_kernel(const float* __restrict__ x,
                         const float* __restrict__ rbf,
                         const int*   __restrict__ idx_i,
                         const float* __restrict__ Wb,   // [128, 6]
                         float* __restrict__ agg,        // [N, 128]
                         int E) {
    const int t    = blockIdx.x * blockDim.x + threadIdx.x;
    const int e    = t >> 5;
    const int lane = t & 31;
    if (e >= E) return;

    const float* rp = rbf + (size_t)e * NUM_RADIAL;
    float r0 = rp[0], r1 = rp[1], r2 = rp[2], r3 = rp[3], r4 = rp[4], r5 = rp[5];

    const int    cbase = lane * 4;
    const float4 xv    = *(const float4*)(x + (size_t)e * HIDDEN + cbase);
    const float  xs[4] = {xv.x, xv.y, xv.z, xv.w};

    const int node = idx_i[e];
    float* dst = agg + (size_t)node * HIDDEN + cbase;

#pragma unroll
    for (int j = 0; j < 4; ++j) {
        const float* w = Wb + (size_t)(cbase + j) * NUM_RADIAL;
        float b = r0 * w[0] + r1 * w[1] + r2 * w[2]
                + r3 * w[3] + r4 * w[4] + r5 * w[5];
        atomicAdd(dst + j, b * xs[j]);   // global_atomic_add_f32
    }
}

// ---------------------------------------------------------------------------
// WMMA fragment loads per documented CDNA5 VGPR layouts (CDNA5 ISA)
// ---------------------------------------------------------------------------

// 16-bit A 16x32: half=0 lanes hold K = kBase+{0..7,16..23},
// half=1 lanes hold K = kBase+{8..15,24..31}; row M = rowBase + (lane&15).
__device__ __forceinline__ v16bf load_a_frag(const __bf16* __restrict__ row,
                                             int kBase, int half) {
    v8bf lo = *(const v8bf*)(row + kBase +      half * 8);   // ds_load_b128
    v8bf hi = *(const v8bf*)(row + kBase + 16 + half * 8);   // ds_load_b128
    return __builtin_shufflevector(lo, hi, 0,1,2,3,4,5,6,7,8,9,10,11,12,13,14,15);
}

// 16-bit B 32x16: column N = colBase + (lane&15); half=0 lanes hold
// K = kBase+{0..7,16..23}, half=1 lanes hold K = kBase+{8..15,24..31}.
__device__ __forceinline__ v16bf load_b_frag(const __bf16* __restrict__ wbuf,
                                             int kin, int colBase, int kBase,
                                             int lane) {
    const __bf16* wp = wbuf + (size_t)(colBase + (lane & 15)) * kin
                            + kBase + (lane >> 4) * 8;
    v8bf lo = *(const v8bf*)(wp);        // ds_load_b128
    v8bf hi = *(const v8bf*)(wp + 16);   // ds_load_b128
    return __builtin_shufflevector(lo, hi, 0,1,2,3,4,5,6,7,8,9,10,11,12,13,14,15);
}

// Cooperative weight staging: GLOBAL_LOAD_ASYNC_TO_LDS_B128 DMA (no VGPR
// round-trip, tracked by ASYNCcnt).  Low 32 bits of a flat shared pointer
// are the LDS byte offset (aperture rules, CDNA5 ISA).
__device__ __forceinline__ void stage_weights_async(__bf16* __restrict__ wbuf,
                                                    const __bf16* __restrict__ src,
                                                    int elems, int tid) {
    const unsigned lds_base = (unsigned)(uintptr_t)wbuf;
    for (int i = tid * 8; i < elems; i += 256 * 8) {
        unsigned lds  = lds_base + (unsigned)i * 2u;
        unsigned goff = (unsigned)i * 2u;
        asm volatile("global_load_async_to_lds_b128 %0, %1, %2"
                     :: "v"(lds), "v"(goff), "s"(src)
                     : "memory");
    }
    asm volatile("s_wait_asynccnt 0x0" ::: "memory");
}

// One dense layer on a wave-private 32-row strip (two 16-row WMMA strips,
// each B fragment feeds two WMMAs).  LDS -> LDS.  ct fully unrolled so all
// ds_store epilogue addresses fold into DS offset immediates.
template <int KIN, int NOUT, bool HAS_BIAS, bool SILU>
__device__ __forceinline__ void mlp_layer(__bf16* __restrict__ act,
                                          const __bf16* __restrict__ wbuf,
                                          const float* __restrict__ bias,
                                          int rowBase, int lane) {
    const int l16  = lane & 15;
    const int half = lane >> 4;
    const __bf16* ar0 = act + (size_t)(rowBase +      l16) * LDS_STRIDE;
    const __bf16* ar1 = act + (size_t)(rowBase + 16 + l16) * LDS_STRIDE;

    v16bf a0[KIN / 32], a1[KIN / 32];
#pragma unroll
    for (int kc = 0; kc < KIN / 32; ++kc) {
        a0[kc] = load_a_frag(ar0, kc * 32, half);
        a1[kc] = load_a_frag(ar1, kc * 32, half);
    }

    const int m0 = rowBase + half * 8;
    __bf16* outp = act + (size_t)m0 * LDS_STRIDE + l16;   // constant-offset base

#pragma unroll
    for (int ct = 0; ct < NOUT / 16; ++ct) {
        const int colBase = ct * 16;
        v8f c0 = {}, c1 = {};
#pragma unroll
        for (int kc = 0; kc < KIN / 32; ++kc) {
            v16bf b = load_b_frag(wbuf, KIN, colBase, kc * 32, lane);
            c0 = __builtin_amdgcn_wmma_f32_16x16x32_bf16(
                     false, a0[kc], false, b, (short)0, c0, false, false);
            c1 = __builtin_amdgcn_wmma_f32_16x16x32_bf16(
                     false, a1[kc], false, b, (short)0, c1, false, false);
        }
        float bv = 0.0f;
        if (HAS_BIAS) bv = bias[colBase + l16];
#pragma unroll
        for (int r = 0; r < 8; ++r) {
            float v0 = c0[r], v1 = c1[r];
            if (HAS_BIAS) { v0 += bv; v1 += bv; }
            if (SILU) {
                v0 = v0 / (1.0f + __expf(-v0));
                v1 = v1 / (1.0f + __expf(-v1));
            }
            outp[(r     ) * LDS_STRIDE + colBase] = (__bf16)v0;   // ds_store_b16 offset:imm
            outp[(r + 16) * LDS_STRIDE + colBase] = (__bf16)v1;
        }
    }
}

// Final 256->128 layer: f32 straight to global output.
__device__ __forceinline__ void final_layer(const __bf16* __restrict__ act,
                                            const __bf16* __restrict__ wbuf,
                                            float* __restrict__ out,
                                            int tile0, int N, int rowBase, int lane) {
    const int l16  = lane & 15;
    const int half = lane >> 4;
    const __bf16* ar0 = act + (size_t)(rowBase +      l16) * LDS_STRIDE;
    const __bf16* ar1 = act + (size_t)(rowBase + 16 + l16) * LDS_STRIDE;

    v16bf a0[OUT_EMB / 32], a1[OUT_EMB / 32];
#pragma unroll
    for (int kc = 0; kc < OUT_EMB / 32; ++kc) {
        a0[kc] = load_a_frag(ar0, kc * 32, half);
        a1[kc] = load_a_frag(ar1, kc * 32, half);
    }

    const int m0 = rowBase + half * 8;
#pragma unroll
    for (int ct = 0; ct < OUT_CH / 16; ++ct) {
        const int colBase = ct * 16;
        v8f c0 = {}, c1 = {};
#pragma unroll
        for (int kc = 0; kc < OUT_EMB / 32; ++kc) {
            v16bf b = load_b_frag(wbuf, OUT_EMB, colBase, kc * 32, lane);
            c0 = __builtin_amdgcn_wmma_f32_16x16x32_bf16(
                     false, a0[kc], false, b, (short)0, c0, false, false);
            c1 = __builtin_amdgcn_wmma_f32_16x16x32_bf16(
                     false, a1[kc], false, b, (short)0, c1, false, false);
        }
        const int n = colBase + l16;
#pragma unroll
        for (int r = 0; r < 8; ++r) {
            int g0 = tile0 + m0 + r;
            int g1 = g0 + 16;
            if (g0 < N) out[(size_t)g0 * OUT_CH + n] = c0[r];
            if (g1 < N) out[(size_t)g1 * OUT_CH + n] = c1[r];
        }
    }
}

// ---------------------------------------------------------------------------
// Kernel 2: node-phase MLP.  8 waves / block, 256 nodes / block.
// Dynamic LDS (256 KB of the WGP's 320 KB): 128 KB bf16 activations +
// 128 KB staged bf16 weights (async-DMA refilled per layer).
// ---------------------------------------------------------------------------
__global__ __launch_bounds__(256)
void node_mlp_kernel(const float*  __restrict__ agg,
                     const __bf16* __restrict__ wblob,   // bf16 weights in ws
                     const float*  __restrict__ b0,
                     const float*  __restrict__ b1,
                     const float*  __restrict__ b2,
                     float* __restrict__ out, int N) {
    extern __shared__ __bf16 smem[];
    __bf16* act  = smem;                          // TILE_M * 256 bf16 = 128 KB
    __bf16* wbuf = smem + TILE_M * LDS_STRIDE;    // up to 256*256 bf16 = 128 KB

    const int tid   = threadIdx.x;
    const int tile0 = blockIdx.x * TILE_M;

    // Stage activation tile (f32 -> bf16, zero-pad rows past N) + proj weights
    for (int i = tid; i < TILE_M * HIDDEN; i += 256) {
        int m = i >> 7, col = i & (HIDDEN - 1);
        int g = tile0 + m;
        float v = (g < N) ? agg[(size_t)g * HIDDEN + col] : 0.0f;
        act[(size_t)m * LDS_STRIDE + col] = (__bf16)v;
    }
    stage_weights_async(wbuf, wblob + OFF_PROJ, OUT_EMB * HIDDEN, tid);
    __syncthreads();

    const int lane    = tid & 31;
    const int rowBase = (tid >> 5) * 32;          // 32 wave-private rows

    mlp_layer<HIDDEN, OUT_EMB, false, false>(act, wbuf, nullptr, rowBase, lane);

    // Three identical 256->256 SiLU layers: one inlined copy in a runtime loop
    for (int l = 0; l < 3; ++l) {
        const float* bias = (l == 0) ? b0 : ((l == 1) ? b1 : b2);
        __syncthreads();                           // everyone done reading wbuf
        stage_weights_async(wbuf, wblob + OFF_W0 + l * (OUT_EMB * OUT_EMB),
                            OUT_EMB * OUT_EMB, tid);
        __syncthreads();
        mlp_layer<OUT_EMB, OUT_EMB, true, true>(act, wbuf, bias, rowBase, lane);
    }

    __syncthreads();
    stage_weights_async(wbuf, wblob + OFF_FIN, OUT_CH * OUT_EMB, tid);
    __syncthreads();
    final_layer(act, wbuf, out, tile0, N, rowBase, lane);
}

// ---------------------------------------------------------------------------
extern "C" void kernel_launch(void* const* d_in, const int* in_sizes, int n_in,
                              void* d_out, int out_size, void* d_ws, size_t ws_size,
                              hipStream_t stream) {
    const float* x     = (const float*)d_in[0];
    const float* rbf   = (const float*)d_in[1];
    const int*   idx_i = (const int*)  d_in[2];
    // d_in[3] = num_nodes scalar (N derived from out_size)
    const float* Wb    = (const float*)d_in[4];
    const float* Wproj = (const float*)d_in[5];
    const float* Wfin  = (const float*)d_in[6];
    const float* W0    = (const float*)d_in[7];
    const float* b0    = (const float*)d_in[8];
    const float* W1    = (const float*)d_in[9];
    const float* b1    = (const float*)d_in[10];
    const float* W2    = (const float*)d_in[11];
    const float* b2    = (const float*)d_in[12];

    const int E = in_sizes[2];
    const int N = out_size / OUT_CH;

    // workspace layout: [ agg f32 N*128 | bf16 weight blob ]
    float*  agg   = (float*)d_ws;
    const int aggElems = N * HIDDEN;
    __bf16* wblob = (__bf16*)((char*)d_ws + (size_t)aggElems * sizeof(float));

    zero_f32_kernel<<<(aggElems + 255) / 256, 256, 0, stream>>>(agg, aggElems);

    // one-shot weight conversion f32 -> bf16
    cvt_bf16_kernel<<<(OUT_EMB * HIDDEN  + 255) / 256, 256, 0, stream>>>(Wproj, wblob + OFF_PROJ, OUT_EMB * HIDDEN);
    cvt_bf16_kernel<<<(OUT_EMB * OUT_EMB + 255) / 256, 256, 0, stream>>>(W0,    wblob + OFF_W0,   OUT_EMB * OUT_EMB);
    cvt_bf16_kernel<<<(OUT_EMB * OUT_EMB + 255) / 256, 256, 0, stream>>>(W1,    wblob + OFF_W1,   OUT_EMB * OUT_EMB);
    cvt_bf16_kernel<<<(OUT_EMB * OUT_EMB + 255) / 256, 256, 0, stream>>>(W2,    wblob + OFF_W2,   OUT_EMB * OUT_EMB);
    cvt_bf16_kernel<<<(OUT_CH  * OUT_EMB + 255) / 256, 256, 0, stream>>>(Wfin,  wblob + OFF_FIN,  OUT_CH  * OUT_EMB);

    const long long edgeThreads = (long long)E * 32;
    edge_scatter_kernel<<<(int)((edgeThreads + 255) / 256), 256, 0, stream>>>(
        x, rbf, idx_i, Wb, agg, E);

    const size_t smemBytes = (size_t)(TILE_M * LDS_STRIDE + OUT_EMB * OUT_EMB) * sizeof(__bf16);
    node_mlp_kernel<<<(N + TILE_M - 1) / TILE_M, 256, smemBytes, stream>>>(
        agg, wblob, b0, b1, b2, (float*)d_out, N);
}
